// RNNGAN_85813446574429
// MI455X (gfx1250) — matrix operand order repaired and checked by reading, not compiled, not run on hardware
//
#include <hip/hip_runtime.h>
#include <hip/hip_bf16.h>
#include <stdint.h>
#include <stddef.h>

// ---------------------------------------------------------------------------
// Problem constants (match reference)
// ---------------------------------------------------------------------------
#define B_    128
#define T_    1024
#define H_    400
#define G4    1600          // 4*H
#define F_IN  56            // 40 random + 16 conditioning
#define KA0   480           // padded K for layer0 A = [h0(400) | x_t(56) | 0-pad(24)]
#define KA1   800           // K for layer1 A = [h1(400) | h0(400)]
#define KC0   15            // 480/32 k-chunks
#define KC1   25            // 800/32 k-chunks
#define NT    100           // 1600/16 gate-column tiles
#define BT_   25            // hidden-column tiles per gate (400/16)
#define NWAVES 16
#define BLK   512

typedef __attribute__((ext_vector_type(16))) __bf16        v16bf;
typedef __attribute__((ext_vector_type(8)))  float         v8f;
typedef __attribute__((ext_vector_type(4)))  unsigned int  v4u;

// ---------------------------------------------------------------------------
// Helpers
// ---------------------------------------------------------------------------
__device__ __forceinline__ unsigned short f32_to_bf16_bits(float f) {
  union { float f; unsigned u; } in; in.f = f;
  unsigned u = in.u;
  unsigned r = u + 0x7FFFu + ((u >> 16) & 1u);   // round-to-nearest-even
  return (unsigned short)(r >> 16);
}
__device__ __forceinline__ __bf16 f32_to_bf16(float f) {
  union { unsigned short s; __bf16 b; } o; o.s = f32_to_bf16_bits(f);
  return o.b;
}
__device__ __forceinline__ float sigmoidf_(float x) {
  return 1.0f / (1.0f + __expf(-x));
}

// A-fragment (16x32 bf16, M=batch rows, K=hidden cols) from LDS row-major A.
// ISA layout: lanes 0-15 hold row M=lane, K = kb+{0..7} and kb+{16..23};
//             lanes 16-31 hold row M=lane-16, K = kb+{8..15} and kb+{24..31}.
__device__ __forceinline__ v16bf load_a_frag(const __bf16* A, int stride, int kb, int lane) {
  const int row = lane & 15;
  const int off = (lane & 16) >> 1;              // 0 or 8
  const __bf16* p = A + row * stride + kb + off; // 16B aligned by construction
  union { v4u u[2]; v16bf v; } t;
  t.u[0] = *(const v4u*)(p);
  t.u[1] = *(const v4u*)(p + 16);
  return t.v;
}

// B-fragment from pre-packed weights: block of 32 lanes x 16 halves, lane-major.
__device__ __forceinline__ v16bf load_b_frag(const __bf16* Wp, int block, int lane) {
  const v16bf* p = (const v16bf*)Wp;
  return p[(size_t)block * 32 + lane];           // 32B contiguous per lane
}

// ---------------------------------------------------------------------------
// Kernel 1: pack fp32 weights -> bf16 B-fragment-ordered blocks.
//   W0p: [Wh0 (400 rows); Wx0 (56 rows); zeros (24 rows)] -> 100 ct x 15 kc blocks
//   W1p: [Wh1 (400 rows); Wx1 (400 rows)]                 -> 100 ct x 25 kc blocks
// Block layout: for block (ct,kc), element (lane,j):
//   n = ct*16 + (lane&15);  k = kc*32 + 8*(lane>>4) + (j<8 ? j : 8+j)
// ---------------------------------------------------------------------------
__global__ __launch_bounds__(256) void pack_weights(
    const float* __restrict__ Wx0, const float* __restrict__ Wh0,
    const float* __restrict__ Wx1, const float* __restrict__ Wh1,
    unsigned short* __restrict__ W0p, unsigned short* __restrict__ W1p)
{
  const int N0 = NT * KC0 * 512;
  const int N1 = NT * KC1 * 512;
  int e = blockIdx.x * 256 + threadIdx.x;
  if (e < N0) {
    int block = e >> 9, w = e & 511;
    int lane = w >> 4, j = w & 15;
    int ct = block / KC0, kc = block - ct * KC0;
    int n = ct * 16 + (lane & 15);
    int k = kc * 32 + 8 * (lane >> 4) + ((j < 8) ? j : 8 + j);
    float v = 0.0f;
    if (k < H_)              v = Wh0[k * G4 + n];
    else if (k < H_ + F_IN)  v = Wx0[(k - H_) * G4 + n];
    W0p[e] = f32_to_bf16_bits(v);
  } else if (e < N0 + N1) {
    int e1 = e - N0;
    int block = e1 >> 9, w = e1 & 511;
    int lane = w >> 4, j = w & 15;
    int ct = block / KC1, kc = block - ct * KC1;
    int n = ct * 16 + (lane & 15);
    int k = kc * 32 + 8 * (lane >> 4) + ((j < 8) ? j : 8 + j);
    float v = (k < H_) ? Wh1[k * G4 + n] : Wx1[(k - H_) * G4 + n];
    W1p[e1] = f32_to_bf16_bits(v);
  }
}

// ---------------------------------------------------------------------------
// Kernel 2: persistent 2-layer LSTM. One workgroup per 16-row batch tile.
// 512 threads = 16 waves; wave owns hidden tiles bt = wave, wave+16 (<25) and
// computes all four gates for those columns -> cell update stays in registers.
// ---------------------------------------------------------------------------
__device__ __forceinline__ void load_x_tile(
    const float* __restrict__ ri, const float* __restrict__ cond,
    __bf16* A0, int bTile, int t, int tid)
{
  for (int idx = tid; idx < 16 * F_IN; idx += BLK) {
    int row = idx / F_IN, col = idx - row * F_IN;
    int b = bTile * 16 + row;
    float v = (col < 40)
        ? ri[((size_t)b * T_ + t) * 40 + col]
        : cond[((size_t)b * T_ + t) * 16 + (col - 40)];
    A0[row * KA0 + H_ + col] = f32_to_bf16(v);
  }
}

__global__ __launch_bounds__(BLK) void lstm_persistent(
    const float* __restrict__ ri, const float* __restrict__ cond,
    const unsigned short* __restrict__ W0p_raw,
    const unsigned short* __restrict__ W1p_raw,
    const float* __restrict__ b0, const float* __restrict__ b1,
    float* __restrict__ hs)
{
  __shared__ __bf16 A0[16 * KA0];   // [h0 | x_t | pad]   15 KB
  __shared__ __bf16 A1[16 * KA1];   // [h1 | h0]          25 KB
  const __bf16* W0p = (const __bf16*)W0p_raw;
  const __bf16* W1p = (const __bf16*)W1p_raw;

  const int tid   = threadIdx.x;
  const int lane  = tid & 31;
  const int wave  = tid >> 5;
  const int bTile = blockIdx.x;     // 0..7

  // Warm L2 with the recurrent weights (they stay resident all 1024 steps).
  __builtin_prefetch(W0p_raw + (size_t)tid * 64, 0, 3);
  __builtin_prefetch(W1p_raw + (size_t)tid * 64, 0, 3);

  // h0 = h1 = 0, zero the K padding, load x_0.
  for (int i = tid; i < 16 * KA0; i += BLK) A0[i] = f32_to_bf16(0.0f);
  for (int i = tid; i < 16 * KA1; i += BLK) A1[i] = f32_to_bf16(0.0f);
  load_x_tile(ri, cond, A0, bTile, 0, tid);
  __syncthreads();

  // Per-wave persistent state: c registers + biases for up to 2 hidden tiles.
  float c0r[2][8], c1r[2][8];
  float bs0[2][4], bs1[2][4];
#pragma unroll
  for (int ii = 0; ii < 2; ++ii) {
    int bt = wave + ii * NWAVES;
    int btc = (bt < BT_) ? bt : 0;
    int nc = btc * 16 + (lane & 15);
#pragma unroll
    for (int r = 0; r < 8; ++r) { c0r[ii][r] = 0.0f; c1r[ii][r] = 0.0f; }
#pragma unroll
    for (int g = 0; g < 4; ++g) {
      bs0[ii][g] = b0[g * H_ + nc];
      bs1[ii][g] = b1[g * H_ + nc];
    }
  }

  for (int t = 0; t < T_; ++t) {
    // ---- Phase 1: gates0 = [h0 | x_t] @ W0p (per-wave: 4 gate tiles) ------
    v8f acc0[2][4];
#pragma unroll
    for (int ii = 0; ii < 2; ++ii) {
      int bt = wave + ii * NWAVES;
      if (bt >= BT_) continue;
#pragma unroll
      for (int g = 0; g < 4; ++g)
#pragma unroll
        for (int r = 0; r < 8; ++r) acc0[ii][g][r] = 0.0f;
      for (int kc = 0; kc < KC0; ++kc) {
        v16bf a = load_a_frag(A0, KA0, kc * 32, lane);
#pragma unroll
        for (int g = 0; g < 4; ++g) {
          v16bf bm = load_b_frag(W0p, (g * BT_ + bt) * KC0 + kc, lane);
          acc0[ii][g] = __builtin_amdgcn_wmma_f32_16x16x32_bf16(
              false, a, false, bm, (short)0, acc0[ii][g], false, false);
        }
      }
    }
    __syncthreads();

    // ---- Phase 2: cell0; h0_new -> A0[:, :400] and A1[:, 400:800] ---------
#pragma unroll
    for (int ii = 0; ii < 2; ++ii) {
      int bt = wave + ii * NWAVES;
      if (bt >= BT_) continue;
      int hc = bt * 16 + (lane & 15);
#pragma unroll
      for (int r = 0; r < 8; ++r) {
        int m = r + ((lane & 16) >> 1);
        float iv = sigmoidf_(acc0[ii][0][r] + bs0[ii][0]);
        float fv = sigmoidf_(acc0[ii][1][r] + bs0[ii][1]);
        float gv = tanhf    (acc0[ii][2][r] + bs0[ii][2]);
        float ov = sigmoidf_(acc0[ii][3][r] + bs0[ii][3]);
        float c  = fv * c0r[ii][r] + iv * gv;
        c0r[ii][r] = c;
        float h  = ov * tanhf(c);
        __bf16 hb = f32_to_bf16(h);
        A0[m * KA0 + hc]       = hb;   // next step's layer-0 A
        A1[m * KA1 + H_ + hc]  = hb;   // this step's layer-1 A ([h1|h0])
      }
    }
    __syncthreads();

    // ---- Phase 3: gates1 = [h1 | h0] @ W1p --------------------------------
    v8f acc1[2][4];
#pragma unroll
    for (int ii = 0; ii < 2; ++ii) {
      int bt = wave + ii * NWAVES;
      if (bt >= BT_) continue;
#pragma unroll
      for (int g = 0; g < 4; ++g)
#pragma unroll
        for (int r = 0; r < 8; ++r) acc1[ii][g][r] = 0.0f;
      for (int kc = 0; kc < KC1; ++kc) {
        v16bf a = load_a_frag(A1, KA1, kc * 32, lane);
#pragma unroll
        for (int g = 0; g < 4; ++g) {
          v16bf bm = load_b_frag(W1p, (g * BT_ + bt) * KC1 + kc, lane);
          acc1[ii][g] = __builtin_amdgcn_wmma_f32_16x16x32_bf16(
              false, a, false, bm, (short)0, acc1[ii][g], false, false);
        }
      }
    }
    __syncthreads();

    // ---- Phase 4: cell1; h1_new -> A1[:, :400] + hs global; prefetch x ----
#pragma unroll
    for (int ii = 0; ii < 2; ++ii) {
      int bt = wave + ii * NWAVES;
      if (bt >= BT_) continue;
      int hc = bt * 16 + (lane & 15);
#pragma unroll
      for (int r = 0; r < 8; ++r) {
        int m = r + ((lane & 16) >> 1);
        float iv = sigmoidf_(acc1[ii][0][r] + bs1[ii][0]);
        float fv = sigmoidf_(acc1[ii][1][r] + bs1[ii][1]);
        float gv = tanhf    (acc1[ii][2][r] + bs1[ii][2]);
        float ov = sigmoidf_(acc1[ii][3][r] + bs1[ii][3]);
        float c  = fv * c1r[ii][r] + iv * gv;
        c1r[ii][r] = c;
        float h  = ov * tanhf(c);
        A1[m * KA1 + hc] = f32_to_bf16(h);
        hs[((size_t)(bTile * 16 + m) * T_ + t) * H_ + hc] = h;
      }
    }
    if (t + 1 < T_) load_x_tile(ri, cond, A0, bTile, t + 1, tid);
    __syncthreads();
  }
}

// ---------------------------------------------------------------------------
// Kernel 3: output dense  out[row, :4] = hs[row, :400] @ Wd + bd.
// One wave per row (coalesced, memory-bound), shuffle reduction.
// ---------------------------------------------------------------------------
__global__ __launch_bounds__(256) void dense_out(
    const float* __restrict__ hs, const float* __restrict__ Wd,
    const float* __restrict__ bd, float* __restrict__ out)
{
  int row  = blockIdx.x * 8 + (threadIdx.x >> 5);
  int lane = threadIdx.x & 31;
  if (row >= B_ * T_) return;
  const float* h = hs + (size_t)row * H_;
  float a0 = 0.f, a1 = 0.f, a2 = 0.f, a3 = 0.f;
  for (int k = lane; k < H_; k += 32) {
    float x = h[k];
    const float* w = Wd + k * 4;
    a0 += x * w[0]; a1 += x * w[1]; a2 += x * w[2]; a3 += x * w[3];
  }
#pragma unroll
  for (int off = 16; off > 0; off >>= 1) {
    a0 += __shfl_xor(a0, off, 32);
    a1 += __shfl_xor(a1, off, 32);
    a2 += __shfl_xor(a2, off, 32);
    a3 += __shfl_xor(a3, off, 32);
  }
  if (lane == 0) {
    out[(size_t)row * 4 + 0] = a0 + bd[0];
    out[(size_t)row * 4 + 1] = a1 + bd[1];
    out[(size_t)row * 4 + 2] = a2 + bd[2];
    out[(size_t)row * 4 + 3] = a3 + bd[3];
  }
}

// ---------------------------------------------------------------------------
// Host entry
// ---------------------------------------------------------------------------
extern "C" void kernel_launch(void* const* d_in, const int* in_sizes, int n_in,
                              void* d_out, int out_size, void* d_ws, size_t ws_size,
                              hipStream_t stream) {
  (void)in_sizes; (void)n_in; (void)out_size; (void)ws_size;
  const float* ri   = (const float*)d_in[0];   // random_input  [B,T,40]
  const float* cond = (const float*)d_in[1];   // conditioning  [B,T,16]
  const float* Wx0  = (const float*)d_in[2];   // [56,1600]
  const float* Wh0  = (const float*)d_in[3];   // [400,1600]
  const float* b0   = (const float*)d_in[4];   // [1600]
  const float* Wx1  = (const float*)d_in[5];   // [400,1600]
  const float* Wh1  = (const float*)d_in[6];   // [400,1600]
  const float* b1   = (const float*)d_in[7];   // [1600]
  const float* Wd   = (const float*)d_in[8];   // [400,4]
  const float* bd   = (const float*)d_in[9];   // [4]
  float* out = (float*)d_out;

  // Workspace layout (bytes, 256-aligned):
  //   [0,          1536000)  W0p  (100*15*512 bf16)
  //   [1536000,    4096000)  W1p  (100*25*512 bf16)
  //   [4096000, 213811200)   hs   (128*1024*400 f32)
  char* ws = (char*)d_ws;
  unsigned short* W0p = (unsigned short*)(ws);
  unsigned short* W1p = (unsigned short*)(ws + 1536000);
  float*          hs  = (float*)(ws + 4096000);

  const int packTotal = NT * KC0 * 512 + NT * KC1 * 512;   // 2,048,000
  pack_weights<<<(packTotal + 255) / 256, 256, 0, stream>>>(Wx0, Wh0, Wx1, Wh1, W0p, W1p);
  lstm_persistent<<<B_ / 16, BLK, 0, stream>>>(ri, cond, W0p, W1p, b0, b1, hs);
  dense_out<<<(B_ * T_ + 7) / 8, 256, 0, stream>>>(hs, Wd, bd, out);
}
